// MultiheadAttention_70033736728852
// MI455X (gfx1250) — matrix operand order repaired and checked by reading, not compiled, not run on hardware
//
#include <hip/hip_runtime.h>
#include <math.h>

// ---------------------------------------------------------------------------
// CDNA5 (gfx1250) bf16 WMMA multihead attention.
//   D = A x B + C via v_wmma_f32_16x16x32_bf16 (wave32, 16x16 tiles, K=32).
// Fragment layouts per CDNA5 ISA §7.12.2:
//   A (16x32 bf16): lane m(0..15) row m, K={0..7,16..23}; lane m+16: K={8..15,24..31}
//   B (32x16 bf16): lane n(0..15) col n, K={0..7,16..23} (2/VGPR); lane n+16: K={8..15,24..31}
//   C/D (16x16 f32): lane n(0..15) col n, VGPR v = row v; lane n+16: row v+8
// ---------------------------------------------------------------------------

typedef __attribute__((ext_vector_type(16))) __bf16 bfrag;  // 8 VGPRs
typedef __attribute__((ext_vector_type(8)))  float  v8f;    // 8 VGPRs

union FragU {
  bfrag f;
  uint4 q[2];
  unsigned short s[16];
};

__device__ inline unsigned short f2bf(float x) {
  unsigned int u = __float_as_uint(x);
  u += 0x7FFFu + ((u >> 16) & 1u);   // round-to-nearest-even
  return (unsigned short)(u >> 16);
}
__device__ inline float bf2f(unsigned short h) {
  return __uint_as_float(((unsigned int)h) << 16);
}

__device__ inline v8f wmma_bf16(bfrag a, bfrag b, v8f c) {
  return __builtin_amdgcn_wmma_f32_16x16x32_bf16(
      /*neg_a=*/false, a, /*neg_b=*/false, b,
      /*c_mod=*/(short)0, c, /*reuse_a=*/false, /*reuse_b=*/false);
}

// A fragment: 16x32 tile of row-major [.,ld] bf16 at (row0, col0).
__device__ inline bfrag load_a_frag(const unsigned short* src, int ld, int row0, int col0) {
  const int l  = threadIdx.x & 31;
  const int m  = l & 15;
  const int hf = l >> 4;
  const unsigned short* p = src + (size_t)(row0 + m) * (size_t)ld + col0 + hf * 8;
  FragU u;
  u.q[0] = *(const uint4*)(p);        // K = hf*8 .. hf*8+7
  u.q[1] = *(const uint4*)(p + 16);   // K = 16+hf*8 ..
  return u.f;
}

// B fragment where source rows ARE the N dimension (K contiguous): B(k,n)=src[(n0+n)*ld + c0+k].
__device__ inline bfrag load_b_ncontig(const unsigned short* src, int ld, int n0, int c0) {
  const int l  = threadIdx.x & 31;
  const int n  = l & 15;
  const int kb = (l >> 4) * 8;
  const unsigned short* p = src + (size_t)(n0 + n) * (size_t)ld + c0 + kb;
  FragU u;
  u.q[0] = *(const uint4*)(p);        // K = kb .. kb+7
  u.q[1] = *(const uint4*)(p + 16);   // K = 16+kb .. 16+kb+7
  return u.f;
}

// A fragment from a 16x32 LDS tile (row stride 32 elements).
__device__ inline bfrag load_a_lds(const unsigned short* src) {
  const int l  = threadIdx.x & 31;
  const int m  = l & 15;
  const int hf = l >> 4;
  const unsigned short* p = src + m * 32 + hf * 8;
  FragU u;
  u.q[0] = *(const uint4*)(p);
  u.q[1] = *(const uint4*)(p + 16);
  return u.f;
}

// ---------------------------------------------------------------------------
// Elementwise f32 -> bf16
// ---------------------------------------------------------------------------
__global__ void k_f32_to_bf16(const float* __restrict__ src, unsigned short* __restrict__ dst, int n) {
  int i = blockIdx.x * blockDim.x + threadIdx.x;
  if (i < n) dst[i] = f2bf(src[i]);
}

// ---------------------------------------------------------------------------
// V transpose: V [B*T, C] (col = h*128+d)  ->  Vt [(b*16+h)*128 + d, T]
// LDS-tiled 32x32 so both global sides are coalesced.
// ---------------------------------------------------------------------------
__global__ __launch_bounds__(256) void k_transpose_v(const unsigned short* __restrict__ V,
                                                     unsigned short* __restrict__ Vt) {
  const int T = 2048, C = 2048;
  __shared__ unsigned short tile[32][33];  // +1 pad to dodge bank conflicts
  const int t0 = (int)(blockIdx.x & 63) * 32;        // T/32 = 64
  const int d0 = (int)((blockIdx.x >> 6) & 3) * 32;  // HD/32 = 4
  const int bh = (int)(blockIdx.x >> 8);             // B*H = 64
  const int b  = bh >> 4, h = bh & 15;
  const int tx = threadIdx.x & 31, ty = threadIdx.x >> 5;  // 32 x 8
#pragma unroll
  for (int r = 0; r < 32; r += 8)
    tile[ty + r][tx] = V[(size_t)(b * T + t0 + ty + r) * C + h * 128 + d0 + tx];
  __syncthreads();
#pragma unroll
  for (int r = 0; r < 32; r += 8)
    Vt[(size_t)(bh * 128 + d0 + ty + r) * T + t0 + tx] = tile[tx][ty + r];
}

// ---------------------------------------------------------------------------
// GEMM: out[m,n] = sum_k A[m,k] * W[n,k]   (A: [M,K] bf16, W: [N,K] bf16)
// Each wave computes a 32x64 output tile; 8 v_wmma per K-step of 32.
// 2-deep register software pipeline: fragments for step kk+32 are loaded
// before the WMMAs for step kk issue, so LOADcnt waits overlap matrix ops.
// ---------------------------------------------------------------------------
template <bool OUT_F32>
__global__ __launch_bounds__(256) void k_gemm_nt(const unsigned short* __restrict__ A,
                                                 const unsigned short* __restrict__ W,
                                                 int M, int N, int K,
                                                 unsigned short* __restrict__ outb,
                                                 float* __restrict__ outf) {
  const int wid    = blockIdx.x * (blockDim.x >> 5) + (threadIdx.x >> 5);
  const int ntiles = N / 64;
  const int mt = wid / ntiles;
  const int nt = wid % ntiles;
  if (mt * 32 >= M) return;
  const int m0 = mt * 32, n0 = nt * 64;

  v8f acc[2][4];
#pragma unroll
  for (int i = 0; i < 2; ++i)
#pragma unroll
    for (int j = 0; j < 4; ++j) acc[i][j] = (v8f){0.f, 0.f, 0.f, 0.f, 0.f, 0.f, 0.f, 0.f};

  // prologue: fragments for kk = 0
  bfrag a0 = load_a_frag(A, K, m0, 0);
  bfrag a1 = load_a_frag(A, K, m0 + 16, 0);
  bfrag b0 = load_b_ncontig(W, K, n0,      0);
  bfrag b1 = load_b_ncontig(W, K, n0 + 16, 0);
  bfrag b2 = load_b_ncontig(W, K, n0 + 32, 0);
  bfrag b3 = load_b_ncontig(W, K, n0 + 48, 0);

  for (int kk = 0; kk < K; kk += 32) {
    bfrag na0 = {}, na1 = {}, nb0 = {}, nb1 = {}, nb2 = {}, nb3 = {};
    const int kn = kk + 32;
    if (kn < K) {  // issue next panel's loads before consuming current fragments
      na0 = load_a_frag(A, K, m0, kn);
      na1 = load_a_frag(A, K, m0 + 16, kn);
      nb0 = load_b_ncontig(W, K, n0,      kn);
      nb1 = load_b_ncontig(W, K, n0 + 16, kn);
      nb2 = load_b_ncontig(W, K, n0 + 32, kn);
      nb3 = load_b_ncontig(W, K, n0 + 48, kn);
    }
    acc[0][0] = wmma_bf16(a0, b0, acc[0][0]);
    acc[1][0] = wmma_bf16(a1, b0, acc[1][0]);
    acc[0][1] = wmma_bf16(a0, b1, acc[0][1]);
    acc[1][1] = wmma_bf16(a1, b1, acc[1][1]);
    acc[0][2] = wmma_bf16(a0, b2, acc[0][2]);
    acc[1][2] = wmma_bf16(a1, b2, acc[1][2]);
    acc[0][3] = wmma_bf16(a0, b3, acc[0][3]);
    acc[1][3] = wmma_bf16(a1, b3, acc[1][3]);
    a0 = na0; a1 = na1; b0 = nb0; b1 = nb1; b2 = nb2; b3 = nb3;
  }

  const int l  = threadIdx.x & 31;
  const int n  = l & 15;
  const int hf = l >> 4;
#pragma unroll
  for (int i = 0; i < 2; ++i)
#pragma unroll
    for (int j = 0; j < 4; ++j)
#pragma unroll
      for (int v = 0; v < 8; ++v) {
        size_t row = (size_t)(m0 + 16 * i + v + 8 * hf);
        size_t col = (size_t)(n0 + 16 * j + n);
        if (OUT_F32) outf[row * N + col] = acc[i][j][v];
        else         outb[row * N + col] = f2bf(acc[i][j][v]);
      }
}

// ---------------------------------------------------------------------------
// RoPE in-place on bf16 Q and K, fold 1/sqrt(HD) into Q.
// Layout [B*T, C], column = h*128 + d. Thread handles pair (d, d+64).
// ---------------------------------------------------------------------------
__global__ void k_rope(unsigned short* __restrict__ Q, unsigned short* __restrict__ K, int total) {
  int idx = blockIdx.x * blockDim.x + threadIdx.x;
  if (idx >= total) return;
  const int i = idx & 63;           // rotary pair index (HD/2 = 64)
  const int h = (idx >> 6) & 15;    // head
  const int t = (idx >> 10) & 2047; // position
  const int b = idx >> 21;
  const size_t off = ((size_t)(b * 2048 + t)) * 2048 + h * 128;

  const float inv = __expf(-(float)i * (9.210340371976184f / 64.0f)); // 10000^(-i/64)
  const float ang = (float)t * inv;
  const float c = __cosf(ang), s = __sinf(ang);
  const float qs = 0.08838834764831845f;  // 1/sqrt(128), score scale folded into Q

  float q1 = bf2f(Q[off + i]), q2 = bf2f(Q[off + i + 64]);
  Q[off + i]      = f2bf((q1 * c - q2 * s) * qs);
  Q[off + i + 64] = f2bf((q2 * c + q1 * s) * qs);

  float k1 = bf2f(K[off + i]), k2 = bf2f(K[off + i + 64]);
  K[off + i]      = f2bf(k1 * c - k2 * s);
  K[off + i + 64] = f2bf(k2 * c + k1 * s);
}

// ---------------------------------------------------------------------------
// Flash attention: one wave per (b, h, 16-row q tile). K-blocks of 32 columns.
// All 8 K fragments load in one clause burst, then 8 S-WMMAs back-to-back;
// V fragments load from the pre-transposed Vt (two b128 per fragment, same
// pattern as K) while softmax runs, then 8 PV-WMMAs. Online softmax in
// C-layout with shfl_xor row reductions; P staged through LDS (C-layout ->
// A-layout). Causal mask applied analytically.
// ---------------------------------------------------------------------------
__global__ __launch_bounds__(128) void k_flash(const unsigned short* __restrict__ Q,
                                               const unsigned short* __restrict__ Km,
                                               const unsigned short* __restrict__ Vt,
                                               unsigned short* __restrict__ Y) {
  const int T = 2048, C = 2048, HD = 128;
  __shared__ __align__(16) unsigned short Pst[4][16 * 32];  // one 16x32 P tile per wave

  const int wiw = threadIdx.x >> 5;
  const int wid = blockIdx.x * 4 + wiw;   // B*H*(T/16) = 8192 waves
  if (wid >= 4 * 16 * (T / 16)) return;
  const int qt = wid & 127;
  const int h  = (wid >> 7) & 15;
  const int b  = wid >> 11;
  const int t0q = qt * 16;
  const int bh128 = (b * 16 + h) * HD;    // Vt row base for this (b,h)

  const int l  = threadIdx.x & 31;
  const int n  = l & 15;
  const int hf = l >> 4;

  // Q fragments: 16 rows x 128 d, as 4 A-fragments (K chunks of 32)
  bfrag qf[4];
#pragma unroll
  for (int c = 0; c < 4; ++c)
    qf[c] = load_a_frag(Q, C, b * T + t0q, h * HD + c * 32);

  v8f o[8];
  float mrow[8], lrow[8];
#pragma unroll
  for (int dt = 0; dt < 8; ++dt) o[dt] = (v8f){0.f, 0.f, 0.f, 0.f, 0.f, 0.f, 0.f, 0.f};
#pragma unroll
  for (int v = 0; v < 8; ++v) { mrow[v] = -INFINITY; lrow[v] = 0.f; }

  const int nkb = (t0q + 47) / 32;  // blocks with any unmasked column
  for (int kb = 0; kb < nkb; ++kb) {
    const int t0k = kb * 32;

    // ---- S = Q @ K^T : burst-load all 8 K fragments, then 8 WMMAs ----
    bfrag kf[8];
#pragma unroll
    for (int nt2 = 0; nt2 < 2; ++nt2)
#pragma unroll
      for (int c = 0; c < 4; ++c)
        kf[nt2 * 4 + c] = load_b_ncontig(Km, C, b * T + t0k + nt2 * 16, h * HD + c * 32);

    v8f s[2];
    s[0] = (v8f){0.f, 0.f, 0.f, 0.f, 0.f, 0.f, 0.f, 0.f};
    s[1] = (v8f){0.f, 0.f, 0.f, 0.f, 0.f, 0.f, 0.f, 0.f};
#pragma unroll
    for (int nt2 = 0; nt2 < 2; ++nt2)
#pragma unroll
      for (int c = 0; c < 4; ++c)
        s[nt2] = wmma_bf16(qf[c], kf[nt2 * 4 + c], s[nt2]);

    // ---- V fragments from Vt (K-contiguous rows = d) load while softmax runs ----
    bfrag vf[8];
#pragma unroll
    for (int dt = 0; dt < 8; ++dt)
      vf[dt] = load_b_ncontig(Vt, T, bh128 + dt * 16, t0k);

    // Causal mask: column > row -> -inf  (C layout: col = t0k + nt2*16 + n, row = t0q + v + 8*hf)
#pragma unroll
    for (int v = 0; v < 8; ++v) {
      const int grow = t0q + v + 8 * hf;
      if (t0k + n      > grow) s[0][v] = -INFINITY;
      if (t0k + 16 + n > grow) s[1][v] = -INFINITY;
    }

    // Online softmax (row reductions across the 16-lane group via shfl_xor)
    unsigned short* pw = Pst[wiw];
#pragma unroll
    for (int v = 0; v < 8; ++v) {
      float mx = fmaxf(s[0][v], s[1][v]);
#pragma unroll
      for (int msk = 8; msk >= 1; msk >>= 1) mx = fmaxf(mx, __shfl_xor(mx, msk, 32));
      const float mnew  = fmaxf(mrow[v], mx);
      const float scale = __expf(mrow[v] - mnew);  // exp(-inf)=0 on first block
      const float p0 = __expf(s[0][v] - mnew);
      const float p1 = __expf(s[1][v] - mnew);
      float rs = p0 + p1;
#pragma unroll
      for (int msk = 8; msk >= 1; msk >>= 1) rs += __shfl_xor(rs, msk, 32);
      lrow[v] = lrow[v] * scale + rs;
      mrow[v] = mnew;
      pw[(v + 8 * hf) * 32 + n]      = f2bf(p0);
      pw[(v + 8 * hf) * 32 + 16 + n] = f2bf(p1);
#pragma unroll
      for (int dt = 0; dt < 8; ++dt) o[dt][v] *= scale;
    }

    // Make the wave's LDS P tile visible before the cross-lane A-fragment read.
    asm volatile("s_wait_dscnt 0x0" ::: "memory");

    bfrag pf = load_a_lds(pw);
#pragma unroll
    for (int dt = 0; dt < 8; ++dt)
      o[dt] = wmma_bf16(pf, vf[dt], o[dt]);
  }

  // Normalize and store y (bf16), layout [B*T, C]
#pragma unroll
  for (int dt = 0; dt < 8; ++dt) {
#pragma unroll
    for (int v = 0; v < 8; ++v) {
      const float val = o[dt][v] / lrow[v];
      const size_t row = (size_t)(b * T + t0q + v + 8 * hf);
      Y[row * C + h * HD + dt * 16 + n] = f2bf(val);
    }
  }
}

// ---------------------------------------------------------------------------
// Launch
// ---------------------------------------------------------------------------
extern "C" void kernel_launch(void* const* d_in, const int* in_sizes, int n_in,
                              void* d_out, int out_size, void* d_ws, size_t ws_size,
                              hipStream_t stream) {
  const int B = 4, T = 2048, C = 2048;
  const int M = B * T;                 // 8192 rows
  const size_t xN = (size_t)M * C;     // 16,777,216
  const size_t wN = (size_t)C * C;     // 4,194,304

  const float* x  = (const float*)d_in[0];
  // d_in[1] = causal mask (handled analytically)
  const float* wq = (const float*)d_in[2];
  const float* wk = (const float*)d_in[3];
  const float* wv = (const float*)d_in[4];
  const float* wo = (const float*)d_in[5];
  float* out = (float*)d_out;

  unsigned short* xb  = (unsigned short*)d_ws;
  unsigned short* wqb = xb + xN;
  unsigned short* wkb = wqb + wN;
  unsigned short* wvb = wkb + wN;
  unsigned short* wob = wvb + wN;
  unsigned short* Qb  = wob + wN;
  unsigned short* Kb  = Qb + xN;
  unsigned short* Vb  = Kb + xN;
  unsigned short* Yb  = Vb + xN;
  unsigned short* Vtb = Yb + xN;   // transposed V [(b*16+h)*128 + d, T]

  // 1) f32 -> bf16
  k_f32_to_bf16<<<(int)((xN + 255) / 256), 256, 0, stream>>>(x, xb, (int)xN);
  k_f32_to_bf16<<<(int)((wN + 255) / 256), 256, 0, stream>>>(wq, wqb, (int)wN);
  k_f32_to_bf16<<<(int)((wN + 255) / 256), 256, 0, stream>>>(wk, wkb, (int)wN);
  k_f32_to_bf16<<<(int)((wN + 255) / 256), 256, 0, stream>>>(wv, wvb, (int)wN);
  k_f32_to_bf16<<<(int)((wN + 255) / 256), 256, 0, stream>>>(wo, wob, (int)wN);

  // 2) projections: Q/K/V = x @ W^T   (8192 waves -> 1024 blocks of 8 waves)
  const int gemm_blocks = (M / 32) * (C / 64) / 8;
  k_gemm_nt<false><<<gemm_blocks, 256, 0, stream>>>(xb, wqb, M, C, C, Qb, nullptr);
  k_gemm_nt<false><<<gemm_blocks, 256, 0, stream>>>(xb, wkb, M, C, C, Kb, nullptr);
  k_gemm_nt<false><<<gemm_blocks, 256, 0, stream>>>(xb, wvb, M, C, C, Vb, nullptr);

  // 3) RoPE (in place, Q also scaled by 1/sqrt(HD)), and V transpose
  const int rope_total = B * T * 16 * 64;
  k_rope<<<(rope_total + 255) / 256, 256, 0, stream>>>(Qb, Kb, rope_total);
  k_transpose_v<<<64 * 4 * 64, 256, 0, stream>>>(Vb, Vtb);

  // 4) causal flash attention -> Yb   (8192 waves -> 2048 blocks of 4 waves)
  k_flash<<<2048, 128, 0, stream>>>(Qb, Kb, Vtb, Yb);

  // 5) output projection: out = y @ wo^T  (f32 output)
  k_gemm_nt<true><<<gemm_blocks, 256, 0, stream>>>(Yb, wob, M, C, C, nullptr, out);
}
